// EfficientHybridCNN_88965952569859
// MI455X (gfx1250) — matrix operand order repaired and checked by reading, not compiled, not run on hardware
//
#include <hip/hip_runtime.h>
#include <hip/hip_bf16.h>

typedef __attribute__((ext_vector_type(16))) _Float16 v16h;
typedef __attribute__((ext_vector_type(8)))  float    v8f;

#define DEV __device__ __forceinline__

// ---------------------------------------------------------------------------
// conv1: direct conv 3->32, 7x7, stride 4, pad 3, + folded BN + ReLU.
// Input x: NCHW f32 (128,3,224,224). Output: NHWC f16 (128,56,56,32).
// Adjacent lanes differ only in oc -> x loads broadcast across the wave.
// ---------------------------------------------------------------------------
__global__ void conv1_kernel(const float* __restrict__ x,
                             const float* __restrict__ w,
                             const float* __restrict__ bias,
                             const float* __restrict__ g,  const float* __restrict__ be,
                             const float* __restrict__ mu, const float* __restrict__ var,
                             _Float16* __restrict__ out)
{
  int idx = blockIdx.x * blockDim.x + threadIdx.x;   // ((b*56+oh)*56+ow)*32+oc
  const int OH = 56, OW = 56;
  if (idx >= 128 * OH * OW * 32) return;
  int oc = idx & 31;
  int t  = idx >> 5;
  int ow = t % OW; t /= OW;
  int oh = t % OH; t /= OH;
  int b  = t;

  float acc = 0.f;
  for (int ic = 0; ic < 3; ++ic) {
    const float* xb = x + ((size_t)(b * 3 + ic)) * 224 * 224;
    const float* wb = w + (oc * 3 + ic) * 49;
    for (int kh = 0; kh < 7; ++kh) {
      int ih = oh * 4 - 3 + kh;
      if (ih < 0 || ih >= 224) continue;
      for (int kw = 0; kw < 7; ++kw) {
        int iw = ow * 4 - 3 + kw;
        if (iw < 0 || iw >= 224) continue;
        acc += xb[ih * 224 + iw] * wb[kh * 7 + kw];
      }
    }
  }
  float scale = g[oc] * rsqrtf(var[oc] + 1e-5f);
  float y = (acc + bias[oc] - mu[oc]) * scale + be[oc];
  out[idx] = (_Float16)fmaxf(y, 0.f);
}

// ---------------------------------------------------------------------------
// Fold BN into an effective bias: ebias = (conv_b - mean)*scale + beta
// ---------------------------------------------------------------------------
__global__ void fold_bias_kernel(const float* __restrict__ cb,
                                 const float* __restrict__ g,  const float* __restrict__ be,
                                 const float* __restrict__ mu, const float* __restrict__ var,
                                 float* __restrict__ ebias, int OC)
{
  int oc = blockIdx.x * blockDim.x + threadIdx.x;
  if (oc >= OC) return;
  float scale = g[oc] * rsqrtf(var[oc] + 1e-5f);
  ebias[oc] = (cb[oc] - mu[oc]) * scale + be[oc];
}

// ---------------------------------------------------------------------------
// Pack conv weights (OC,IC,3,3) f32, scaled by the BN factor, into the WMMA
// A-fragment order (contiguous K: lane half h holds K=16h..16h+15) as f16:
//   wA[(((mt*KT)+kt)*32 + lane)*16 + e]
// GEMM K ordered as (kh, kw, ic) so B fragments are channel-contiguous.
// ---------------------------------------------------------------------------
__global__ void pack_wA_kernel(const float* __restrict__ w,
                               const float* __restrict__ g, const float* __restrict__ var,
                               _Float16* __restrict__ wA, int OC, int IC)
{
  int KT = (9 * IC) / 32;
  int MT = OC / 16;
  int idx = blockIdx.x * blockDim.x + threadIdx.x;
  if (idx >= MT * KT * 32 * 16) return;
  int e    = idx & 15;
  int lane = (idx >> 4) & 31;
  int kt   = (idx >> 9) % KT;
  int mt   = (idx >> 9) / KT;

  int m  = mt * 16 + (lane & 15);
  int hi = lane >> 4;
  int v  = e >> 1, p = e & 1;
  // contiguous K order (matches B): lanes 0-15 K=0..15, lanes 16-31 K=16..31
  int klocal = hi * 16 + v * 2 + p;
  int k  = kt * 32 + klocal;
  int tap = k / IC, ic = k % IC;
  int kh = tap / 3, kw = tap % 3;

  float scale = g[m] * rsqrtf(var[m] + 1e-5f);
  wA[idx] = (_Float16)(w[((m * IC + ic) * 3 + kh) * 3 + kw] * scale);
}

// ---------------------------------------------------------------------------
// conv2/conv3 as implicit GEMM on v_wmma_f32_16x16x32_f16.
//   M = OC, N = B*OH*OW, K = 9*IC (ordered kh,kw,ic), stride 2, pad 1.
// Compile-time geometry: K-loop fully unrolls (9 / 18 iterations), all index
// divisions strength-reduce to shifts/magic multiplies.
// One wave computes a 16(M) x 32(N) tile: the A fragment is loaded once per
// k-step and feeds TWO independent WMMAs (A reuse + latency hiding per the
// 7.12.1 co-exec hazard table). B fragments are single contiguous 32-byte
// channel-vector loads per lane (zero at padding).
// Epilogue: +folded-BN bias, ReLU, f16 NHWC store.
// ---------------------------------------------------------------------------
template<int IC, int OC, int IH, int IW, int OH, int OW>
__global__ void conv_wmma_kernel(const _Float16* __restrict__ in,   // NHWC (B,IH,IW,IC)
                                 const _Float16* __restrict__ wA,   // packed A fragments
                                 const float* __restrict__ ebias,
                                 _Float16* __restrict__ out)        // NHWC (B,OH,OW,OC)
{
  constexpr int KT = (9 * IC) / 32;
  constexpr int MT = OC / 16;
  int wave = (int)((blockIdx.x * blockDim.x + threadIdx.x) >> 5);
  int lane = threadIdx.x & 31;
  int mt = wave % MT;
  int np = wave / MT;                       // pair of adjacent n-tiles
  int hi = lane >> 4;
  int ln = lane & 15;

  // Decode the two output columns this lane contributes to.
  int n0  = np * 32 + ln;
  int n1  = n0 + 16;
  int ow0 = n0 % OW, t0 = n0 / OW, oh0 = t0 % OH, b0 = t0 / OH;
  int ow1 = n1 % OW, t1 = n1 / OW, oh1 = t1 % OH, b1 = t1 / OH;

  v8f acc0 = {};
  v8f acc1 = {};
  const _Float16* wAt = wA + ((size_t)mt * KT) * 32 * 16 + (size_t)lane * 16;

#pragma unroll
  for (int kt = 0; kt < KT; ++kt) {
    v16h a = *(const v16h*)(wAt + (size_t)kt * 32 * 16);

    int k0  = kt * 32 + hi * 16;            // 16 consecutive K values this lane
    int tap = k0 / IC, ic0 = k0 % IC;       // stays within one (kh,kw) tap: 16 | IC
    int kh = tap / 3, kw = tap % 3;

    int ih0 = oh0 * 2 - 1 + kh, iw0 = ow0 * 2 - 1 + kw;
    int ih1 = oh1 * 2 - 1 + kh, iw1 = ow1 * 2 - 1 + kw;

    v16h bf0 = {};
    if (ih0 >= 0 && ih0 < IH && iw0 >= 0 && iw0 < IW)
      bf0 = *(const v16h*)(in + (((size_t)(b0 * IH + ih0)) * IW + iw0) * IC + ic0);
    v16h bf1 = {};
    if (ih1 >= 0 && ih1 < IH && iw1 >= 0 && iw1 < IW)
      bf1 = *(const v16h*)(in + (((size_t)(b1 * IH + ih1)) * IW + iw1) * IC + ic0);

    acc0 = __builtin_amdgcn_wmma_f32_16x16x32_f16(false, a, false, bf0,
                                                  (short)0, acc0, false, false);
    acc1 = __builtin_amdgcn_wmma_f32_16x16x32_f16(false, a, false, bf1,
                                                  (short)0, acc1, false, false);
  }

#pragma unroll
  for (int r = 0; r < 8; ++r) {
    int oc  = mt * 16 + hi * 8 + r;         // C/D layout: lanes 16-31 hold M+8
    float eb = ebias[oc];
    float y0 = fmaxf(acc0[r] + eb, 0.f);
    float y1 = fmaxf(acc1[r] + eb, 0.f);
    out[(((size_t)(b0 * OH + oh0)) * OW + ow0) * OC + oc] = (_Float16)y0;
    out[(((size_t)(b1 * OH + oh1)) * OW + ow1) * OC + oc] = (_Float16)y1;
  }
}

// ---------------------------------------------------------------------------
// AdaptiveAvgPool2d((2,2)) on NHWC f16 (128,14,14,128) -> f (128,512) f32,
// feature index = c*4 + i*2 + j  (matches reshape(B,128,2,7,2,7).mean(3,5))
// ---------------------------------------------------------------------------
__global__ void pool_kernel(const _Float16* __restrict__ h, float* __restrict__ f)
{
  int idx = blockIdx.x * blockDim.x + threadIdx.x;   // (b,c,i,j)
  if (idx >= 128 * 128 * 2 * 2) return;
  int j = idx & 1;
  int i = (idx >> 1) & 1;
  int c = (idx >> 2) & 127;
  int b = idx >> 9;
  float s = 0.f;
  for (int y = 0; y < 7; ++y)
    for (int x = 0; x < 7; ++x)
      s += (float)h[(((size_t)(b * 14 + i * 7 + y)) * 14 + (j * 7 + x)) * 128 + c];
  f[(size_t)b * 512 + c * 4 + i * 2 + j] = s * (1.0f / 49.0f);
}

// ---------------------------------------------------------------------------
// Head: linear 512->4, tanh*pi, 4-qubit statevector sim, FC 4->64->5.
// One thread per batch element; 16 complex amplitudes live in registers.
// Wire 0 is the MSB of the state index (PennyLane convention).
// ---------------------------------------------------------------------------
DEV void apply1q(float* re, float* im, int q,
                 float g00r, float g00i, float g01r, float g01i,
                 float g10r, float g10i, float g11r, float g11i)
{
  int bit = 1 << (3 - q);
#pragma unroll
  for (int i = 0; i < 16; ++i) {
    if (i & bit) continue;
    int j = i | bit;
    float ar = re[i], ai = im[i], br = re[j], bi = im[j];
    re[i] = g00r * ar - g00i * ai + g01r * br - g01i * bi;
    im[i] = g00r * ai + g00i * ar + g01r * bi + g01i * br;
    re[j] = g10r * ar - g10i * ai + g11r * br - g11i * bi;
    im[j] = g10r * ai + g10i * ar + g11r * bi + g11i * br;
  }
}

DEV void cnot(float* re, float* im, int c, int t)
{
  int cb = 1 << (3 - c), tb = 1 << (3 - t);
#pragma unroll
  for (int i = 0; i < 16; ++i) {
    if ((i & cb) && !(i & tb)) {
      int j = i | tb;
      float tr = re[i]; re[i] = re[j]; re[j] = tr;
      float ti = im[i]; im[i] = im[j]; im[j] = ti;
    }
  }
}

__global__ void head_kernel(const float* __restrict__ f,
                            const float* __restrict__ pre_w, const float* __restrict__ pre_b,
                            const float* __restrict__ qw,
                            const float* __restrict__ pw1, const float* __restrict__ pb1,
                            const float* __restrict__ pw2, const float* __restrict__ pb2,
                            float* __restrict__ outp)
{
  int b = blockIdx.x * blockDim.x + threadIdx.x;
  if (b >= 128) return;

  float ang[4];
  for (int o = 0; o < 4; ++o) {
    float s = pre_b[o];
    const float* fb = f + (size_t)b * 512;
    const float* wo = pre_w + o * 512;
    for (int k = 0; k < 512; ++k) s += fb[k] * wo[k];
    ang[o] = tanhf(s) * 3.14159265358979323846f;
  }

  float re[16], im[16];
#pragma unroll
  for (int i = 0; i < 16; ++i) { re[i] = 0.f; im[i] = 0.f; }
  re[0] = 1.f;

  for (int q = 0; q < 4; ++q) {                       // initial RY(angles)
    float c = cosf(ang[q] * 0.5f), s = sinf(ang[q] * 0.5f);
    apply1q(re, im, q, c, 0.f, -s, 0.f, s, 0.f, c, 0.f);
  }
  for (int l = 0; l < 2; ++l) {
    for (int q = 0; q < 4; ++q) {
      float t0 = qw[(l * 4 + q) * 3 + 0];
      float t1 = qw[(l * 4 + q) * 3 + 1];
      float t2 = qw[(l * 4 + q) * 3 + 2];
      float c, s;
      c = cosf(t0 * 0.5f); s = sinf(t0 * 0.5f);       // RX
      apply1q(re, im, q, c, 0.f, 0.f, -s, 0.f, -s, c, 0.f);
      c = cosf(t1 * 0.5f); s = sinf(t1 * 0.5f);       // RY
      apply1q(re, im, q, c, 0.f, -s, 0.f, s, 0.f, c, 0.f);
      c = cosf(t2 * 0.5f); s = sinf(t2 * 0.5f);       // RZ = diag(e^{-it/2}, e^{it/2})
      apply1q(re, im, q, c, -s, 0.f, 0.f, 0.f, 0.f, c, s);
    }
    for (int q = 0; q < 4; ++q) cnot(re, im, q, (q + 1) & 3);   // ring CNOTs
  }

  float qout[4];
  for (int k = 0; k < 4; ++k) {
    float s = 0.f; int bit = 1 << (3 - k);
#pragma unroll
    for (int i = 0; i < 16; ++i) {
      float p = re[i] * re[i] + im[i] * im[i];
      s += (i & bit) ? -p : p;
    }
    qout[k] = s;
  }

  float h1[64];
  for (int o = 0; o < 64; ++o) {
    float s = pb1[o];
    for (int k = 0; k < 4; ++k) s += qout[k] * pw1[o * 4 + k];
    h1[o] = fmaxf(s, 0.f);
  }
  for (int j = 0; j < 5; ++j) {
    float s = pb2[j];
    for (int o = 0; o < 64; ++o) s += h1[o] * pw2[j * 64 + o];
    outp[b * 5 + j] = s;
  }
}

// ---------------------------------------------------------------------------
extern "C" void kernel_launch(void* const* d_in, const int* in_sizes, int n_in,
                              void* d_out, int out_size, void* d_ws, size_t ws_size,
                              hipStream_t stream)
{
  (void)in_sizes; (void)n_in; (void)out_size; (void)ws_size;

  const float* x    = (const float*)d_in[0];
  const float* c1w  = (const float*)d_in[1];
  const float* c1b  = (const float*)d_in[2];
  const float* bn1g = (const float*)d_in[3];
  const float* bn1b = (const float*)d_in[4];
  const float* bn1m = (const float*)d_in[5];
  const float* bn1v = (const float*)d_in[6];
  const float* c2w  = (const float*)d_in[7];
  const float* c2b  = (const float*)d_in[8];
  const float* bn2g = (const float*)d_in[9];
  const float* bn2b = (const float*)d_in[10];
  const float* bn2m = (const float*)d_in[11];
  const float* bn2v = (const float*)d_in[12];
  const float* c3w  = (const float*)d_in[13];
  const float* c3b  = (const float*)d_in[14];
  const float* bn3g = (const float*)d_in[15];
  const float* bn3b = (const float*)d_in[16];
  const float* bn3m = (const float*)d_in[17];
  const float* bn3v = (const float*)d_in[18];
  const float* prew = (const float*)d_in[19];
  const float* preb = (const float*)d_in[20];
  const float* qw   = (const float*)d_in[21];
  const float* pw1  = (const float*)d_in[22];
  const float* pb1  = (const float*)d_in[23];
  const float* pw2  = (const float*)d_in[24];
  const float* pb2  = (const float*)d_in[25];
  float* outp = (float*)d_out;

  // Workspace carving (256-B aligned so v16h (32-B) loads stay aligned).
  size_t off = 0;
  auto carve = [&](size_t bytes) -> void* {
    off = (off + 255) & ~(size_t)255;
    void* p = (char*)d_ws + off;
    off += bytes;
    return p;
  };
  _Float16* h1o = (_Float16*)carve((size_t)128 * 56 * 56 * 32 * sizeof(_Float16));
  _Float16* h2o = (_Float16*)carve((size_t)128 * 28 * 28 * 64 * sizeof(_Float16));
  _Float16* h3o = (_Float16*)carve((size_t)128 * 14 * 14 * 128 * sizeof(_Float16));
  _Float16* wA2 = (_Float16*)carve((size_t)4 * 9 * 32 * 16 * sizeof(_Float16));
  _Float16* wA3 = (_Float16*)carve((size_t)8 * 18 * 32 * 16 * sizeof(_Float16));
  float*    eb2 = (float*)carve(64 * sizeof(float));
  float*    eb3 = (float*)carve(128 * sizeof(float));
  float*    fpl = (float*)carve((size_t)128 * 512 * sizeof(float));

  // conv1 (direct VALU) -> NHWC f16
  {
    int total = 128 * 56 * 56 * 32;
    conv1_kernel<<<(total + 255) / 256, 256, 0, stream>>>(
        x, c1w, c1b, bn1g, bn1b, bn1m, bn1v, h1o);
  }
  // conv2 prep + WMMA GEMM: M=64, N=100352, K=288
  fold_bias_kernel<<<1, 64, 0, stream>>>(c2b, bn2g, bn2b, bn2m, bn2v, eb2, 64);
  pack_wA_kernel<<<(4 * 9 * 32 * 16 + 255) / 256, 256, 0, stream>>>(c2w, bn2g, bn2v, wA2, 64, 32);
  // waves = (100352/32) n-pairs * 4 m-tiles = 12544; 8 waves/block -> 1568 blocks
  conv_wmma_kernel<32, 64, 56, 56, 28, 28><<<1568, 256, 0, stream>>>(h1o, wA2, eb2, h2o);

  // conv3 prep + WMMA GEMM: M=128, N=25088, K=576
  fold_bias_kernel<<<1, 128, 0, stream>>>(c3b, bn3g, bn3b, bn3m, bn3v, eb3, 128);
  pack_wA_kernel<<<(8 * 18 * 32 * 16 + 255) / 256, 256, 0, stream>>>(c3w, bn3g, bn3v, wA3, 128, 64);
  // waves = (25088/32) n-pairs * 8 m-tiles = 6272; 8 waves/block -> 784 blocks
  conv_wmma_kernel<64, 128, 28, 28, 14, 14><<<784, 256, 0, stream>>>(h2o, wA3, eb3, h3o);

  // pool + head
  pool_kernel<<<256, 256, 0, stream>>>(h3o, fpl);
  head_kernel<<<1, 128, 0, stream>>>(fpl, prew, preb, qw, pw1, pb1, pw2, pb2, outp);
}
